// QLSTM_65481071407628
// MI455X (gfx1250) — compile-verified, matrix operand order checked by reading, not run on hardware
//
#include <hip/hip_runtime.h>

#define S_LEN 1024
#define BATCH 64
#define DI 256
#define DH 256
#define NH 4
#define DHH 64
#define K_DIM 256

typedef _Float16 v16h __attribute__((ext_vector_type(16)));
typedef float    v8f  __attribute__((ext_vector_type(8)));
typedef unsigned int u32x4 __attribute__((ext_vector_type(4)));
typedef int          i32x8 __attribute__((ext_vector_type(8)));
typedef int          i32x4 __attribute__((ext_vector_type(4)));

union AFrag { v16h v; _Float16 e[16]; unsigned int u[8]; };
union CFrag { v8f v; float f[8]; };

// Load an 8-dword f16 fragment as two 16B-aligned uint4s (-> b128 loads).
// dwbase points at dword (row*256 + kb)/2; halves are 8 dwords apart.
__device__ __forceinline__ void load_frag8(AFrag& f, const unsigned int* p,
                                           int dwbase)
{
  const uint4 q0 = *(const uint4*)(p + dwbase);
  const uint4 q1 = *(const uint4*)(p + dwbase + 8);
  f.u[0] = q0.x; f.u[1] = q0.y; f.u[2] = q0.z; f.u[3] = q0.w;
  f.u[4] = q1.x; f.u[5] = q1.y; f.u[6] = q1.z; f.u[7] = q1.w;
}

// ---------------------------------------------------------------------------
// Prep: transpose + f32->f16 convert weights into [N][K] layout, concat biases.
// ---------------------------------------------------------------------------
__global__ __launch_bounds__(256) void prep_kernel(
    const float* __restrict__ Wf, const float* __restrict__ Wi,
    const float* __restrict__ Wg, const float* __restrict__ Wo,
    const float* __restrict__ bf, const float* __restrict__ bi,
    const float* __restrict__ bg, const float* __restrict__ bo,
    const float* __restrict__ inw, const float* __restrict__ outw,
    _Float16* __restrict__ Wx_t, _Float16* __restrict__ Whh_t,
    _Float16* __restrict__ inproj_t, _Float16* __restrict__ outproj_t,
    float* __restrict__ bias4)
{
  int tid = blockIdx.x * blockDim.x + threadIdx.x;
  if (tid < 1024 * 256) {
    int ng = tid >> 8, k = tid & 255;
    int g = ng >> 8, n = ng & 255;
    const float* W = (g == 0) ? Wf : (g == 1) ? Wi : (g == 2) ? Wg : Wo;
    Wx_t [ng * 256 + k] = (_Float16)W[k * 256 + n];          // rows 0..255 (x part)
    Whh_t[ng * 256 + k] = (_Float16)W[(256 + k) * 256 + n];  // rows 256..511 (h part)
    if (k == 0) {
      const float* bb = (g == 0) ? bf : (g == 1) ? bi : (g == 2) ? bg : bo;
      bias4[ng] = bb[n];
    }
  } else if (tid < 1024 * 256 + 768 * 256) {
    int r = tid - 1024 * 256;
    int n3 = r >> 8, k = r & 255;
    inproj_t[n3 * 256 + k] = (_Float16)inw[k * 768 + n3];
  } else if (tid < 1024 * 256 + 768 * 256 + 256 * 256) {
    int r = tid - (1024 * 256 + 768 * 256);
    int n = r >> 8, k = r & 255;
    outproj_t[n * 256 + k] = (_Float16)outw[k * 256 + n];
  }
}

// elementwise f32 -> f16 (grid sized exactly: n % 1024 == 0)
__global__ __launch_bounds__(1024) void cvt16_kernel(
    const float* __restrict__ src, _Float16* __restrict__ dst)
{
  int i = blockIdx.x * 1024 + threadIdx.x;
  dst[i] = (_Float16)src[i];
}

// ---------------------------------------------------------------------------
// WMMA GEMM: Out[M][N] = A16[M][256] @ Wt^T + bias (+Res).
// A16: f16 row-major (pre-converted).  Wt: f16 [N][256].
// Each wave: one 16x32 output (2 N-tiles sharing the A-fragment), 8 waves/blk.
// Grids are exact multiples so EXEC is all-1s at every WMMA.
// ---------------------------------------------------------------------------
__global__ __launch_bounds__(256) void gemm_f16w(
    const _Float16* __restrict__ A16,
    const _Float16* __restrict__ Wt,
    const float* __restrict__ bias,
    const float* __restrict__ Res,      // optional residual [M][N]
    _Float16* __restrict__ Out16,       // used if out16 != 0
    float* __restrict__ Out32,          // used if out16 == 0
    int N, int nhalf, int out16)        // nhalf = (N/16)/2
{
  const int lane = threadIdx.x & 31;
  const int wave = threadIdx.x >> 5;
  const int pair = blockIdx.x * 8 + wave;
  const int mt = pair / nhalf, nt0 = pair % nhalf;
  const int m0 = mt * 16;
  const int row = m0 + (lane & 15);          // A-fragment M row for this lane
  const int nA  = nt0 * 16 + (lane & 15);    // tile 0 N col
  const int nB  = nA + nhalf * 16;           // tile 1 N col
  const int khi = (lane >> 4) * 8;
  const unsigned int* A32 = (const unsigned int*)A16;
  const unsigned int* W32 = (const unsigned int*)Wt;
  const int adw = (row * K_DIM) >> 1;        // dword bases
  const int bdw0 = (nA * K_DIM) >> 1;
  const int bdw1 = (nB * K_DIM) >> 1;

  v8f acc0 = {}, acc1 = {};
  #pragma unroll
  for (int kc = 0; kc < 8; ++kc) {
    AFrag a, b0, b1;
    const int kdw = (kc * 32 + khi) >> 1;    // dword offset of this k-chunk
    load_frag8(a,  A32, adw  + kdw);
    load_frag8(b0, W32, bdw0 + kdw);
    load_frag8(b1, W32, bdw1 + kdw);
    acc0 = __builtin_amdgcn_wmma_f32_16x16x32_f16(
        false, a.v, false, b0.v, (short)0, acc0, false, false);
    acc1 = __builtin_amdgcn_wmma_f32_16x16x32_f16(
        false, a.v, false, b1.v, (short)0, acc1, false, false);
  }

  const float bn0 = bias[nA];
  const float bn1 = bias[nB];
  #pragma unroll
  for (int v = 0; v < 8; ++v) {
    const int m = m0 + v + ((lane >> 4) << 3);
    float v0 = acc0[v] + bn0;
    float v1 = acc1[v] + bn1;
    if (Res) { v0 += Res[m * N + nA]; v1 += Res[m * N + nB]; }
    if (out16) {
      Out16[(size_t)m * N + nA] = (_Float16)v0;
      Out16[(size_t)m * N + nB] = (_Float16)v1;
    } else {
      Out32[(size_t)m * N + nA] = v0;
      Out32[(size_t)m * N + nB] = v1;
    }
  }
}

// ---------------------------------------------------------------------------
// Sequential LSTM on one WGP (32 wave32 waves).  h(f16)/c(f32) in LDS.
// Per step: the next step's 128KB pre-gate slice is staged LDS-side by the
// Tensor Data Mover (TENSOR_LOAD_TO_LDS, TENSORcnt) while WMMAs run.
// LDS: 64K cstate + 32K hbuf (static) + 128K pgbuf (dynamic) = 224KB.
// ---------------------------------------------------------------------------
__device__ __forceinline__ void tdm_load_slice(const _Float16* gsrc,
                                               unsigned lds_off)
{
  // 1-D tile: 16384 elements x 8B = 128KB, groups 2/3 zero (<=2-D tensor).
  const unsigned long long ga = (unsigned long long)gsrc;
  u32x4 g0;
  g0.x = 1u;                                        // count=1, user mode
  g0.y = lds_off;                                   // lds_addr (bytes)
  g0.z = (unsigned)(ga & 0xFFFFFFFFull);            // global_addr[31:0]
  g0.w = (unsigned)((ga >> 32) & 0x1FFFFFFull)      // global_addr[56:32]
       | 0x80000000u;                               // type=2 ("image")
  i32x8 g1;
  g1[0] = 0x00030000;           // data_size=3 (8B), no multicast/pad/iterate
  g1[1] = (int)(16384u << 16);  // tensor_dim0[15:0] -> bits 63:48
  g1[2] = 0x00010000;           // tensor_dim0 hi = 0, tensor_dim1 = 1
  g1[3] = (int)(16384u << 16);  // tile_dim0 = 16384 -> bits 127:112
  g1[4] = 0;                    // tile_dim1 = 0 (unused), tile_dim2 = 0
  g1[5] = 16384;                // tensor_dim0_stride lo
  g1[6] = 0;
  g1[7] = 0;
  i32x4 gz4 = {0, 0, 0, 0};
  i32x8 gz8 = {0, 0, 0, 0, 0, 0, 0, 0};
  // 6-arg toolchain variant: (g0, g1, g2, g3, g4, cpol)
  __builtin_amdgcn_tensor_load_to_lds(g0, g1, gz4, gz4, gz8, 0);
}

__global__ __launch_bounds__(1024) void lstm_kernel(
    const _Float16* __restrict__ pregates,   // [S*64][1024] f16 (x@Wx + bias)
    const _Float16* __restrict__ Whh_t,      // [1024][256] f16
    float* __restrict__ Hseq,                // [S*64][256] f32
    _Float16* __restrict__ Hseq16,           // [S*64][256] f16
    float* __restrict__ hx_out,              // [64][256]
    float* __restrict__ cx_out)              // [64][256]
{
  __shared__ float    cstate[BATCH * DH];    // 64 KB
  __shared__ _Float16 hbuf [BATCH * DH];     // 32 KB
  extern __shared__ _Float16 pgbuf[];        // 128 KB dynamic: [64][1024]

  const int tid = threadIdx.x;
  for (int i = tid; i < BATCH * DH; i += 1024) {
    cstate[i] = 0.0f;
    hbuf[i]   = (_Float16)0.0f;
  }

  const int lane = tid & 31;
  const int wave = tid >> 5;
  const unsigned pg_lds = (unsigned)(size_t)&pgbuf[0];   // LDS byte offset

  // stage step 0 pre-gates via TDM
  if (wave == 0) {
    tdm_load_slice(pregates, pg_lds);
    __builtin_amdgcn_s_wait_tensorcnt(0);
  }
  __syncthreads();

  const int mt   = wave >> 3;                // 0..3  -> batch tile
  const int dt0  = wave & 7;                 // dtiles dt0 and dt0+8
  const int m0   = mt * 16;
  const int mrow = m0 + (lane & 15);
  const int khi  = (lane >> 4) * 8;
  const int ncol = lane & 15;
  const unsigned int* hb32 = (const unsigned int*)hbuf;
  const unsigned int* Wh32 = (const unsigned int*)Whh_t;
  const int adw = (mrow * DH) >> 1;          // hbuf dword base

  for (int s = 0; s < S_LEN; ++s) {
    CFrag acc[2][4];

    // seed accumulators from LDS-staged x-part pre-activations
    #pragma unroll
    for (int t = 0; t < 2; ++t) {
      const int d0 = (dt0 + t * 8) * 16;
      #pragma unroll
      for (int g = 0; g < 4; ++g) {
        #pragma unroll
        for (int v = 0; v < 8; ++v) {
          const int m = m0 + v + ((lane >> 4) << 3);
          acc[t][g].f[v] = (float)pgbuf[m * 1024 + g * 256 + d0 + ncol];
        }
      }
    }
    __syncthreads();   // all seed reads done; pgbuf may be overwritten

    // kick TDM for step s+1 while this step's WMMAs execute
    if (wave == 0 && s + 1 < S_LEN) {
      tdm_load_slice(pregates + (size_t)(s + 1) * BATCH * 1024, pg_lds);
    }

    // h @ Whh : 8 k-chunks x (2 dtiles x 4 gates) WMMA, A-frag reused 8x
    #pragma unroll 1
    for (int kc = 0; kc < 8; ++kc) {
      AFrag a;
      const int kdw = (kc * 32 + khi) >> 1;
      load_frag8(a, hb32, adw + kdw);        // ds_load_b128 x2
      #pragma unroll
      for (int t = 0; t < 2; ++t) {
        const int d0 = (dt0 + t * 8) * 16;
        #pragma unroll
        for (int g = 0; g < 4; ++g) {
          AFrag b;
          const int nb = g * 256 + d0 + ncol;
          load_frag8(b, Wh32, ((nb * K_DIM) >> 1) + kdw);
          acc[t][g].v = __builtin_amdgcn_wmma_f32_16x16x32_f16(
              false, a.v, false, b.v, (short)0, acc[t][g].v, false, false);
        }
      }
    }
    __syncthreads();   // all GEMM reads of hbuf complete

    // cell update; each (m,d) element owned by exactly one lane/vgpr
    #pragma unroll
    for (int t = 0; t < 2; ++t) {
      const int d0 = (dt0 + t * 8) * 16;
      #pragma unroll
      for (int v = 0; v < 8; ++v) {
        const int m = m0 + v + ((lane >> 4) << 3);
        const int d = d0 + ncol;
        const float fg = 1.0f / (1.0f + __expf(-acc[t][0].f[v]));
        const float ig = 1.0f / (1.0f + __expf(-acc[t][1].f[v]));
        const float gg = tanhf(acc[t][2].f[v]);
        const float og = 1.0f / (1.0f + __expf(-acc[t][3].f[v]));
        const float c  = fg * cstate[m * DH + d] + ig * gg;
        cstate[m * DH + d] = c;
        const float h = og * tanhf(c);
        hbuf[m * DH + d] = (_Float16)h;
        Hseq  [((size_t)s * BATCH + m) * DH + d] = h;
        Hseq16[((size_t)s * BATCH + m) * DH + d] = (_Float16)h;
        if (s == S_LEN - 1) {
          hx_out[m * DH + d] = h;
          cx_out[m * DH + d] = c;
        }
      }
    }
    if (wave == 0) __builtin_amdgcn_s_wait_tensorcnt(0);
    __syncthreads();   // new h visible + pgbuf(s+1) staged
  }
}

// ---------------------------------------------------------------------------
// Batch-axis attention per (s, head): q,k,v are [64][64]; LDS tiles + register
// softmax.  ~3% of total FLOPs, VALU is fine here.  Emits f16 ctx for the
// out-projection GEMM.
// ---------------------------------------------------------------------------
__global__ __launch_bounds__(64) void attn_kernel(
    const _Float16* __restrict__ qkv,   // [S*64][768] f16
    _Float16* __restrict__ ctx16)       // [S*64][256] f16
{
  __shared__ float Qs[64 * 64], Ks[64 * 64], Vs[64 * 64];
  const int s = blockIdx.x >> 2;
  const int h = blockIdx.x & 3;
  const int t = threadIdx.x;

  const _Float16* base = qkv + (size_t)s * BATCH * 768 + h * DHH;
  #pragma unroll 4
  for (int d = 0; d < 64; ++d) {
    Qs[t * 64 + d] = (float)base[t * 768 + d];
    Ks[t * 64 + d] = (float)base[t * 768 + 256 + d];
    Vs[t * 64 + d] = (float)base[t * 768 + 512 + d];
  }
  __syncthreads();

  float sc[64];
  #pragma unroll 4
  for (int j = 0; j < 64; ++j) {
    float dot = 0.0f;
    #pragma unroll
    for (int d = 0; d < 64; ++d) dot += Qs[t * 64 + d] * Ks[j * 64 + d];
    sc[j] = dot * 0.125f;   // 1/sqrt(64)
  }
  float mx = -1e30f;
  #pragma unroll
  for (int j = 0; j < 64; ++j) mx = fmaxf(mx, sc[j]);
  float sum = 0.0f;
  #pragma unroll
  for (int j = 0; j < 64; ++j) { sc[j] = __expf(sc[j] - mx); sum += sc[j]; }
  const float inv = 1.0f / sum;

  _Float16* out = ctx16 + ((size_t)s * BATCH + t) * DH + h * DHH;
  #pragma unroll 4
  for (int d = 0; d < 64; ++d) {
    float a = 0.0f;
    #pragma unroll
    for (int j = 0; j < 64; ++j) a += sc[j] * Vs[j * 64 + d];
    out[d] = (_Float16)(a * inv);
  }
}

// ---------------------------------------------------------------------------
extern "C" void kernel_launch(void* const* d_in, const int* in_sizes, int n_in,
                              void* d_out, int out_size, void* d_ws, size_t ws_size,
                              hipStream_t stream) {
  const float* inputs = (const float*)d_in[0];
  const float* Wf = (const float*)d_in[1];  const float* bf = (const float*)d_in[2];
  const float* Wi = (const float*)d_in[3];  const float* bi = (const float*)d_in[4];
  const float* Wg = (const float*)d_in[5];  const float* bg = (const float*)d_in[6];
  const float* Wo = (const float*)d_in[7];  const float* bo = (const float*)d_in[8];
  const float* in_proj_w  = (const float*)d_in[9];
  const float* in_proj_b  = (const float*)d_in[10];
  const float* out_proj_w = (const float*)d_in[11];
  const float* out_proj_b = (const float*)d_in[12];

  float* fused_out = (float*)d_out;                                 // [S*64][256]
  float* hx_out = fused_out + (size_t)S_LEN * BATCH * DH;           // [64][256]
  float* cx_out = hx_out + BATCH * DH;

  // workspace layout (regions time-share; stream order guarantees safety)
  char* ws = (char*)d_ws;
  _Float16* pregates = (_Float16*)ws;                               // 134,217,728 B
  _Float16* qkv16    = (_Float16*)ws;                               // reuse (100 MB)
  float* Hseq = (float*)(ws + 134217728ull);                        //  67,108,864 B
  char* regR  = ws + 134217728ull + 67108864ull;                    //  67,108,864 B
  _Float16* Xf16   = (_Float16*)regR;                               // steps 1-2
  _Float16* Hseq16 = (_Float16*)(regR + 33554432ull);               // steps 3-4
  _Float16* ctx16  = (_Float16*)regR;                               // steps 5-6
  char* wbase = ws + 268435456ull;
  _Float16* Wx_t      = (_Float16*)wbase;                           // 524,288 B
  _Float16* Whh_t     = (_Float16*)(wbase + 524288);                // 524,288 B
  _Float16* inproj_t  = (_Float16*)(wbase + 1048576);               // 393,216 B
  _Float16* outproj_t = (_Float16*)(wbase + 1441792);               // 131,072 B
  float*    bias4     = (float*)   (wbase + 1572864);               //   4,096 B

  // 1) weight transpose/convert + input f16 copy
  prep_kernel<<<2048, 256, 0, stream>>>(Wf, Wi, Wg, Wo, bf, bi, bg, bo,
                                        in_proj_w, out_proj_w,
                                        Wx_t, Whh_t, inproj_t, outproj_t, bias4);
  cvt16_kernel<<<16384, 1024, 0, stream>>>(inputs, Xf16);

  // 2) x-part of all LSTM gates: [65536,256] @ [256,1024] + bias -> f16
  gemm_f16w<<<16384, 256, 0, stream>>>(Xf16, Wx_t, bias4, nullptr,
                                       pregates, nullptr, 1024, 32, 1);

  // 3) sequential recurrence on one WGP (128KB dynamic LDS for TDM staging)
  lstm_kernel<<<1, 1024, 131072, stream>>>(pregates, Whh_t, Hseq, Hseq16,
                                           hx_out, cx_out);

  // 4) QKV projection: [65536,256] @ [256,768] -> f16 (reuses pregates region)
  gemm_f16w<<<12288, 256, 0, stream>>>(Hseq16, inproj_t, in_proj_b, nullptr,
                                       qkv16, nullptr, 768, 24, 1);

  // 5) batch-axis attention per (s, head)
  attn_kernel<<<S_LEN * NH, 64, 0, stream>>>(qkv16, ctx16);

  // 6) out-projection + residual -> fused output (f32)
  gemm_f16w<<<4096, 256, 0, stream>>>(ctx16, outproj_t, out_proj_b, Hseq,
                                      nullptr, fused_out, 256, 8, 0);
}
